// CausalSelfAttention_84224308674661
// MI455X (gfx1250) — matrix operand order repaired and checked by reading, not compiled, not run on hardware
//
#include <hip/hip_runtime.h>

// ---------------------------------------------------------------------------
// CDNA5 (gfx1250) causal self-attention forward.
// bf16 WMMA (v_wmma_f32_16x16x32_bf16) for QKV GEMM, QK^T, PV, and out-proj.
// Wave32; fragment layouts follow CDNA5 ISA 16-bit A/B/C tables.
// Weights are converted bf16 + transposed once, so GEMM A and B tiles are
// both straight copies staged with GLOBAL_LOAD_ASYNC_TO_LDS_B128 (ASYNCcnt).
// ---------------------------------------------------------------------------

typedef __bf16 bf16;
typedef __attribute__((ext_vector_type(8)))  __bf16 bf16x8;
typedef __attribute__((ext_vector_type(16))) __bf16 bf16x16;
typedef __attribute__((ext_vector_type(8)))  float  f32x8;

union Frag16 { bf16x16 v; bf16x8 h[2]; };

#define WMMA_BF16(a, b, c) \
  __builtin_amdgcn_wmma_f32_16x16x32_bf16(false, (a), false, (b), (short)0, (c), false, false)

#if defined(__gfx1250__) && __has_builtin(__builtin_amdgcn_global_load_async_to_lds_b128)
#define HAVE_ASYNC_LDS 1
typedef int v4i_vs __attribute__((vector_size(16)));
typedef __attribute__((address_space(1))) v4i_vs* as1_v4i;
typedef __attribute__((address_space(3))) v4i_vs* as3_v4i;
__device__ __forceinline__ void async_cpy16(void* lds, const void* g) {
  __builtin_amdgcn_global_load_async_to_lds_b128(
      (as1_v4i)g, (as3_v4i)lds, 0, 0);
}
#define ASYNC_WAIT() asm volatile("s_wait_asynccnt 0x0" ::: "memory")
#else
#define HAVE_ASYNC_LDS 0
__device__ __forceinline__ void async_cpy16(void* lds, const void* g) {
  *(bf16x8*)lds = *(const bf16x8*)g;   // sync fallback
}
#define ASYNC_WAIT()
#endif

#define T_SEQ 2048
#define HD    128
#define NH    16
#define NB    2

// ---------------------------------------------------------------------------
// Elementwise f32 -> bf16 convert (grid-stride).
// ---------------------------------------------------------------------------
__global__ void cvt_f32_bf16(const float* __restrict__ in, bf16* __restrict__ out, long n) {
  long i = (long)blockIdx.x * blockDim.x + threadIdx.x;
  long stride = (long)gridDim.x * blockDim.x;
  for (; i < n; i += stride) out[i] = (bf16)in[i];
}

// ---------------------------------------------------------------------------
// Transposing convert: in[K][N] f32 (row-major) -> out[N][K] bf16.
// LDS-tiled 32x32 so both global sides stay coalesced.
// ---------------------------------------------------------------------------
__global__ __launch_bounds__(256) void cvt_transpose_f32_bf16(
    const float* __restrict__ in, bf16* __restrict__ out, int K, int N) {
  __shared__ float tile[32][33];
  const int kb = blockIdx.y * 32, nb = blockIdx.x * 32;
  const int tx = threadIdx.x, ty = threadIdx.y;   // block (32, 8)
#pragma unroll
  for (int i = 0; i < 32; i += 8)
    tile[ty + i][tx] = in[(size_t)(kb + ty + i) * N + nb + tx];
  __syncthreads();
#pragma unroll
  for (int i = 0; i < 32; i += 8)
    out[(size_t)(nb + ty + i) * K + kb + tx] = (bf16)tile[tx][ty + i];
}

// ---------------------------------------------------------------------------
// bf16 GEMM, f32 accumulate: C[M][N] = A[M][K] * Bt[N][K]^T   (row-major)
// Block tile 128x256, BK=32, 256 threads = 8 waves, wave tile 64x64.
// Double-buffered LDS; both tiles staged via async-to-LDS DMA (straight
// copies: A rows and Bt rows are K-contiguous, matching fragment layout).
// ---------------------------------------------------------------------------
__global__ __launch_bounds__(256) void gemm_bf16(const bf16* __restrict__ A,
                                                 const bf16* __restrict__ Bt,
                                                 float* __restrict__ C,
                                                 int M, int N, int K) {
  __shared__ __align__(16) bf16 As[2][128][32];
  __shared__ __align__(16) bf16 Bs[2][256][32];   // [n][k]

  const int tid  = threadIdx.x;
  const int wave = tid >> 5;
  const int lane = tid & 31;
  const int half = lane >> 4;     // 0: lanes 0-15, 1: lanes 16-31
  const int l16  = lane & 15;
  const int wrow = wave >> 2;     // 0..1 -> 64-row slab
  const int wcol = wave & 3;      // 0..3 -> 64-col slab
  const int m0 = blockIdx.y * 128;
  const int n0 = blockIdx.x * 256;

  f32x8 acc[4][4];
#pragma unroll
  for (int i = 0; i < 4; ++i)
#pragma unroll
    for (int j = 0; j < 4; ++j)
#pragma unroll
      for (int e = 0; e < 8; ++e) acc[i][j][e] = 0.0f;

  const int tmA = tid >> 2;          // 0..63 (+64 for 2nd slab)
  const int tkA = (tid & 3) * 8;     // 0,8,16,24
  const int kA  = half * 8;          // A-fragment lane-half K offset

  const bf16* Aptr = A  + (size_t)(m0 + tmA) * K + tkA;
  const bf16* Bptr = Bt + (size_t)(n0 + tid) * K;
  const size_t A2 = (size_t)64 * K;

  // Stage one K-tile (A: 2 chunks/thread, B: 4 chunks/thread, all async).
  auto stage = [&](int buf, int kb) {
    async_cpy16(&As[buf][tmA][tkA],      Aptr + kb);
    async_cpy16(&As[buf][tmA + 64][tkA], Aptr + kb + A2);
#pragma unroll
    for (int c = 0; c < 4; ++c)
      async_cpy16(&Bs[buf][tid][c * 8], Bptr + kb + c * 8);
  };

  // ---- Prologue: stage tile kb=0 into buffer 0 -----------------------------
  stage(0, 0);
  ASYNC_WAIT();
  __syncthreads();

  int cur = 0;
  for (int kb = 0; kb < K; kb += 32) {
    if (kb + 32 < K) stage(cur ^ 1, kb + 32);   // prefetch next tile (async)

    // ---- Compute on buffer `cur` -------------------------------------------
    Frag16 af[4], bfr[4];
#pragma unroll
    for (int mt = 0; mt < 4; ++mt) {
      const bf16* p = &As[cur][wrow * 64 + mt * 16 + l16][0];
      af[mt].h[0] = *(const bf16x8*)(p + kA);        // K = kA..kA+7
      af[mt].h[1] = *(const bf16x8*)(p + 16 + kA);   // K = 16+kA..
    }
#pragma unroll
    for (int nt = 0; nt < 4; ++nt) {
      const bf16* p = &Bs[cur][wcol * 64 + nt * 16 + l16][0];
      bfr[nt].h[0] = *(const bf16x8*)(p + half * 8);      // K = half*8..
      bfr[nt].h[1] = *(const bf16x8*)(p + half * 8 + 16);
    }
#pragma unroll
    for (int mt = 0; mt < 4; ++mt)
#pragma unroll
      for (int nt = 0; nt < 4; ++nt)
        acc[mt][nt] = WMMA_BF16(af[mt].v, bfr[nt].v, acc[mt][nt]);

    ASYNC_WAIT();       // own prefetches landed
    __syncthreads();    // everyone's landed; buffer flip safe
    cur ^= 1;
  }

  // C layout: VGPR r -> row r (lanes 0-15) / row 8+r (lanes 16-31); col = lane%16.
#pragma unroll
  for (int mt = 0; mt < 4; ++mt)
#pragma unroll
    for (int nt = 0; nt < 4; ++nt)
#pragma unroll
      for (int r = 0; r < 8; ++r) {
        int row = m0 + wrow * 64 + mt * 16 + half * 8 + r;
        int col = n0 + wcol * 64 + nt * 16 + l16;
        C[(size_t)row * N + col] = acc[mt][nt][r];
      }
}

// ---------------------------------------------------------------------------
// RoPE on q,k; split/transpose qkv[B,T,3C] -> Q/K/V bf16 [B,H,T,D].
// One thread per (b,t,h, d-pair).
// ---------------------------------------------------------------------------
__global__ void rope_kernel(const float* __restrict__ qkv,
                            bf16* __restrict__ Qb,
                            bf16* __restrict__ Kb,
                            bf16* __restrict__ Vb) {
  long idx = (long)blockIdx.x * blockDim.x + threadIdx.x;
  const long total = (long)NB * T_SEQ * NH * (HD / 2);
  if (idx >= total) return;
  int dp = idx & 63;                 // pair index 0..63
  int h  = (idx >> 6) & 15;
  int t  = (idx >> 10) & 2047;
  int b  = (int)(idx >> 21);

  size_t row = ((size_t)b * T_SEQ + t) * (3 * NH * HD);
  int col = h * HD + 2 * dp;
  float q0 = qkv[row + col],        q1 = qkv[row + col + 1];
  float k0 = qkv[row + 2048 + col], k1 = qkv[row + 2048 + col + 1];
  float v0 = qkv[row + 4096 + col], v1 = qkv[row + 4096 + col + 1];

  float inv = __powf(10000.0f, -((float)(2 * dp)) / (float)HD);
  float ang = (float)t * inv;
  float s, c;
  __sincosf(ang, &s, &c);

  size_t ob = (((size_t)b * NH + h) * T_SEQ + t) * HD + 2 * dp;
  Qb[ob]     = (bf16)(q0 * c - q1 * s);
  Qb[ob + 1] = (bf16)(q1 * c + q0 * s);
  Kb[ob]     = (bf16)(k0 * c - k1 * s);
  Kb[ob + 1] = (bf16)(k1 * c + k0 * s);
  Vb[ob]     = (bf16)v0;
  Vb[ob + 1] = (bf16)v1;
}

// 16-lane-half cross-lane reductions (wave32; xor masks 1,2,4,8 stay in half).
__device__ inline float redmax16(float v) {
#pragma unroll
  for (int m = 1; m < 16; m <<= 1) v = fmaxf(v, __shfl_xor(v, m, 32));
  return v;
}
__device__ inline float redsum16(float v) {
#pragma unroll
  for (int m = 1; m < 16; m <<= 1) v += __shfl_xor(v, m, 32);
  return v;
}

// ---------------------------------------------------------------------------
// Flash attention: block = (b*H+h, qtile of 64). 4 waves x 16 q-rows.
// K tile [64][128] staged via async LDS-DMA (straight copy).
// V tile staged transposed VT[d][key] through registers.
// P is re-fragmented to A layout through per-wave LDS.
// Output written as bf16 in [B,T,H,D] so it feeds the proj GEMM directly.
// ---------------------------------------------------------------------------
__global__ __launch_bounds__(128) void attn_kernel(const bf16* __restrict__ Q,
                                                   const bf16* __restrict__ K,
                                                   const bf16* __restrict__ V,
                                                   bf16* __restrict__ Y) {
  __shared__ __align__(16) bf16 Ks[64][HD];
  __shared__ __align__(16) bf16 VT[HD][64];
  __shared__ __align__(16) bf16 Ps[4][16][64];

  const int bh  = blockIdx.x;          // b*NH + h
  const int qt  = blockIdx.y;          // query tile of 64
  const int q0  = qt * 64;
  const int tid = threadIdx.x;
  const int wave = tid >> 5, lane = tid & 31;
  const int half = lane >> 4, l16 = lane & 15;
  const int kA = half * 8;
  const size_t base = (size_t)bh * T_SEQ * HD;

  // Staging decomposition: thread -> (srow 0..7, scol 0..120), chunk c adds 8 rows.
  const int srow = tid >> 4;
  const int scol = (tid & 15) * 8;

  // Q fragments: rows q0 + wave*16 + lane%16, 4 K-steps of 32 over head dim.
  Frag16 qf[4];
  {
    const bf16* qp = Q + base + (size_t)(q0 + wave * 16 + l16) * HD;
#pragma unroll
    for (int g = 0; g < 4; ++g) {
      qf[g].h[0] = *(const bf16x8*)(qp + g * 32 + kA);
      qf[g].h[1] = *(const bf16x8*)(qp + g * 32 + 16 + kA);
    }
  }

  f32x8 o[8];
#pragma unroll
  for (int n = 0; n < 8; ++n)
#pragma unroll
    for (int e = 0; e < 8; ++e) o[n][e] = 0.0f;
  float m_i[8], l_i[8];
#pragma unroll
  for (int r = 0; r < 8; ++r) { m_i[r] = -1e30f; l_i[r] = 0.0f; }

  const float scale = 0.08838834764831845f;  // 1/sqrt(128)

  for (int kt = 0; kt <= qt; ++kt) {
    __syncthreads();
    const bf16* kp = K + base + (size_t)kt * 64 * HD + (size_t)srow * HD + scol;
    const bf16* vp = V + base + (size_t)kt * 64 * HD + (size_t)srow * HD + scol;

    // K tile: straight copy, async DMA (overlaps the V transpose below).
#pragma unroll
    for (int c = 0; c < 8; ++c)
      async_cpy16(&Ks[srow + c * 8][scol], kp + (size_t)c * 8 * HD);
    // V tile: batch global loads, then transpose-scatter into VT[d][key].
    bf16x8 vreg[8];
#pragma unroll
    for (int c = 0; c < 8; ++c) vreg[c] = *(const bf16x8*)(vp + (size_t)c * 8 * HD);
#pragma unroll
    for (int c = 0; c < 8; ++c)
#pragma unroll
      for (int j = 0; j < 8; ++j) VT[scol + j][srow + c * 8] = vreg[c][j];
    ASYNC_WAIT();
    __syncthreads();

    // S = Q * K^T  (16q x 64k per wave = 4 accumulators, 4 K-steps each).
    f32x8 s[4];
#pragma unroll
    for (int nt = 0; nt < 4; ++nt) {
      f32x8 a;
#pragma unroll
      for (int e = 0; e < 8; ++e) a[e] = 0.0f;
#pragma unroll
      for (int g = 0; g < 4; ++g) {
        Frag16 bf;
        const bf16* bp = &Ks[nt * 16 + l16][g * 32 + half * 8];
        bf.h[0] = *(const bf16x8*)bp;
        bf.h[1] = *(const bf16x8*)(bp + 16);
        a = WMMA_BF16(qf[g].v, bf.v, a);
      }
      s[nt] = a;
    }

    // Scale + causal mask (diagonal tile only).
#pragma unroll
    for (int nt = 0; nt < 4; ++nt)
#pragma unroll
      for (int r = 0; r < 8; ++r) {
        float val = s[nt][r] * scale;
        if (kt == qt) {
          int kg = kt * 64 + nt * 16 + l16;
          int qg = q0 + wave * 16 + half * 8 + r;
          if (kg > qg) val = -1e30f;
        }
        s[nt][r] = val;
      }

    // Online softmax.
    float mnew[8], alpha[8];
#pragma unroll
    for (int r = 0; r < 8; ++r) {
      float t = fmaxf(fmaxf(s[0][r], s[1][r]), fmaxf(s[2][r], s[3][r]));
      t = redmax16(t);
      mnew[r]  = fmaxf(m_i[r], t);
      alpha[r] = __expf(m_i[r] - mnew[r]);
      m_i[r]   = mnew[r];
    }
#pragma unroll
    for (int n = 0; n < 8; ++n)
#pragma unroll
      for (int r = 0; r < 8; ++r) o[n][r] *= alpha[r];

#pragma unroll
    for (int r = 0; r < 8; ++r) {
      float rs = 0.0f;
#pragma unroll
      for (int nt = 0; nt < 4; ++nt) {
        float p = __expf(s[nt][r] - mnew[r]);
        s[nt][r] = p;
        rs += p;
      }
      l_i[r] = l_i[r] * alpha[r] + redsum16(rs);
    }

    // Re-fragment P (C layout -> A layout) through per-wave LDS.
#pragma unroll
    for (int nt = 0; nt < 4; ++nt)
#pragma unroll
      for (int r = 0; r < 8; ++r)
        Ps[wave][half * 8 + r][nt * 16 + l16] = (bf16)s[nt][r];
    // Same-wave DS ops are in order; compiler inserts s_wait_dscnt.

    Frag16 pf[2];
#pragma unroll
    for (int ks = 0; ks < 2; ++ks) {
      const bf16* pp = &Ps[wave][l16][ks * 32 + kA];
      pf[ks].h[0] = *(const bf16x8*)pp;
      pf[ks].h[1] = *(const bf16x8*)(pp + 16);
    }

    // O += P * V.
#pragma unroll
    for (int n = 0; n < 8; ++n)
#pragma unroll
      for (int ks = 0; ks < 2; ++ks) {
        Frag16 bf;
        const bf16* vv = &VT[n * 16 + l16][ks * 32 + half * 8];
        bf.h[0] = *(const bf16x8*)vv;
        bf.h[1] = *(const bf16x8*)(vv + 16);
        o[n] = WMMA_BF16(pf[ks].v, bf.v, o[n]);
      }
  }

  // Normalize and store bf16 to Y[B,T,H,D] (feeds the projection GEMM).
  const int b = bh >> 4, h = bh & 15;
#pragma unroll
  for (int n = 0; n < 8; ++n)
#pragma unroll
    for (int r = 0; r < 8; ++r) {
      int qg = q0 + wave * 16 + half * 8 + r;
      Y[(((size_t)b * T_SEQ + qg) * NH + h) * HD + n * 16 + l16] =
          (bf16)(o[n][r] / l_i[r]);
    }
}

// ---------------------------------------------------------------------------
// Host-side orchestration.
// ---------------------------------------------------------------------------
extern "C" void kernel_launch(void* const* d_in, const int* in_sizes, int n_in,
                              void* d_out, int out_size, void* d_ws, size_t ws_size,
                              hipStream_t stream) {
  const float* x      = (const float*)d_in[0];
  const float* w_qkv  = (const float*)d_in[1];
  const float* w_proj = (const float*)d_in[2];
  float* out = (float*)d_out;
  char* ws = (char*)d_ws;

  const long BT = (long)NB * T_SEQ;   // 4096
  const long C  = 2048, QKV = 6144;

  size_t off = 0;
  bf16*  x_b   = (bf16*)(ws + off);  off += (size_t)BT * C   * 2;   // 16 MB
  bf16*  wqT_b = (bf16*)(ws + off);  off += (size_t)C  * QKV * 2;   // 24 MB (transposed [N][K])
  bf16*  wpT_b = (bf16*)(ws + off);  off += (size_t)C  * C   * 2;   //  8 MB (transposed [N][K])
  float* qkv   = (float*)(ws + off); off += (size_t)BT * QKV * 4;   // 96 MB
  bf16*  Qb    = (bf16*)(ws + off);  off += (size_t)BT * C * 2;     // 16 MB
  bf16*  Kb    = (bf16*)(ws + off);  off += (size_t)BT * C * 2;     // 16 MB
  bf16*  Vb    = (bf16*)(ws + off);  off += (size_t)BT * C * 2;     // 16 MB
  bf16*  y_b   = (bf16*)(ws + off);  off += (size_t)BT * C * 2;     // 16 MB

  cvt_f32_bf16<<<2048, 256, 0, stream>>>(x, x_b, BT * C);
  cvt_transpose_f32_bf16<<<dim3(192, 64), dim3(32, 8), 0, stream>>>(w_qkv,  wqT_b, 2048, 6144);
  cvt_transpose_f32_bf16<<<dim3(64, 64),  dim3(32, 8), 0, stream>>>(w_proj, wpT_b, 2048, 2048);

  // qkv = x @ w_qkv   [4096,2048] x [2048,6144]  (B passed transposed)
  gemm_bf16<<<dim3(24, 32), 256, 0, stream>>>(x_b, wqT_b, qkv, 4096, 6144, 2048);

  // RoPE + split to Q/K/V bf16 [B,H,T,D]
  rope_kernel<<<16384, 256, 0, stream>>>(qkv, Qb, Kb, Vb);

  // Flash attention -> y bf16 [B,T,H,D]
  attn_kernel<<<dim3(32, 32), 128, 0, stream>>>(Qb, Kb, Vb, y_b);

  // out = y @ w_proj   [4096,2048] x [2048,2048]  (B passed transposed)
  gemm_bf16<<<dim3(8, 32), 256, 0, stream>>>(y_b, wpT_b, out, 4096, 2048, 2048);
}
